// RetiNetLanguageModel_27161373180376
// MI455X (gfx1250) — matrix operand order repaired and checked by reading, not compiled, not run on hardware
//
#include <hip/hip_runtime.h>
#include <hip/hip_bf16.h>

typedef __attribute__((ext_vector_type(16))) __bf16 bf16x16;
typedef __attribute__((ext_vector_type(8)))  __bf16 bf16x8;
typedef __attribute__((ext_vector_type(2)))  __bf16 bf16x2;
typedef __attribute__((ext_vector_type(8)))  float  f32x8;
typedef __attribute__((ext_vector_type(4)))  unsigned int u32x4;
typedef __attribute__((ext_vector_type(8)))  int i32x8;
typedef __attribute__((ext_vector_type(4)))  int i32x4;

#define S_LEN 2048
#define D_DIM 512
#define NHEAD 4
#define VOCAB 32000
#define BATCH 2
#define HD    (NHEAD * D_DIM)   // 2048
#define ROWS  (BATCH * S_LEN)   // 4096
#define KT    64                // GEMM K-tile (two WMMA K-steps)

__device__ __forceinline__ f32x8 wmma_bf16f32(bf16x16 a, bf16x16 b, f32x8 c) {
  return __builtin_amdgcn_wmma_f32_16x16x32_bf16(false, a, false, b, (short)0, c,
                                                 false, false);
}

__device__ __forceinline__ f32x8 f32x8_zero() {
  f32x8 z;
#pragma unroll
  for (int i = 0; i < 8; ++i) z[i] = 0.0f;
  return z;
}

// A fragment (16 rows x 32 K, bf16) from a row-major source with leading dim ld.
__device__ __forceinline__ bf16x16 load_a_frag(const __bf16* base, int ld, int lane) {
  const int lh = lane >> 4, r = lane & 15;
  const __bf16* p = base + (size_t)r * ld + lh * 8;
  bf16x8 lo = *(const bf16x8*)p;
  bf16x8 hi = *(const bf16x8*)(p + 16);
  return __builtin_shufflevector(lo, hi, 0, 1, 2, 3, 4, 5, 6, 7,
                                 8, 9, 10, 11, 12, 13, 14, 15);
}

// B fragment (32 K x 16 N, bf16); source holds N-columns as rows of ld elems.
__device__ __forceinline__ bf16x16 load_b_frag(const __bf16* base, int ld, int lane) {
  const int lh = lane >> 4, r = lane & 15;
  const __bf16* p = base + (size_t)r * ld + lh * 8;
  bf16x8 lo = *(const bf16x8*)p;
  bf16x8 hi = *(const bf16x8*)(p + 16);
  return __builtin_shufflevector(lo, hi, 0, 1, 2, 3, 4, 5, 6, 7,
                                 8, 9, 10, 11, 12, 13, 14, 15);
}

// ---------------------------------------------------------------- TDM
// Issue a 2-D tensor_load_to_lds: tile_x (elements, 2B each) x tile_y rows,
// row stride = stride_elems, from gaddr into LDS byte offset lds_byte_off.
// D# layout per CDNA5 ISA 8.3/8.4 (group0: count/lds/global/type=2;
// group1: data_size=2B, dims/tile/stride). Groups 2/3 zero (<=2D tensor).
// 6-arg builtin: (uint32x4 g0, int32x8 g1, int32x4, int32x4, int32x8, cpol)
__device__ __forceinline__ void tdm_load_2d(unsigned lds_byte_off,
                                            const void* gaddr, int tile_x,
                                            int tile_y, long long stride_elems) {
  unsigned long long ga = (unsigned long long)gaddr;
  u32x4 g0;
  g0[0] = 1u;                                   // count=1 (user descriptor)
  g0[1] = lds_byte_off;                         // lds_addr
  g0[2] = (unsigned)(ga & 0xffffffffu);         // global_addr[31:0]
  g0[3] = (unsigned)((ga >> 32) & 0x01ffffffu)  // global_addr[56:32]
          | (2u << 30);                         // type=2 ("image")
  i32x8 g1;
  g1[0] = 1 << 16;                              // data_size=2B, mask=0
  g1[1] = (tile_x & 0xffff) << 16;              // tensor_dim0[15:0]
  g1[2] = ((tile_x >> 16) & 0xffff) | ((tile_y & 0xffff) << 16);  // d0 hi,d1 lo
  g1[3] = ((tile_y >> 16) & 0xffff) | ((tile_x & 0xffff) << 16);  // d1 hi,tile0
  g1[4] = tile_y & 0xffff;                      // tile_dim1 (tile_dim2=0)
  g1[5] = (int)(stride_elems & 0xffffffffll);   // tensor_dim0_stride[31:0]
  g1[6] = (int)((stride_elems >> 32) & 0xffff); // stride[47:32]
  g1[7] = 0;
  i32x4 z4;
  z4[0] = 0; z4[1] = 0; z4[2] = 0; z4[3] = 0;
  i32x8 z8;
#pragma unroll
  for (int i = 0; i < 8; ++i) z8[i] = 0;
  __builtin_amdgcn_tensor_load_to_lds(g0, g1, z4, z4, z8, 0);
}

__device__ __forceinline__ unsigned lds_off(const void* p) {
  // Low 32 bits of a generic LDS pointer are the in-allocation byte offset.
  return (unsigned)(unsigned long long)p;
}

// ---------------------------------------------------------------- converts
__global__ void cvt_f32_bf16_kernel(const float* __restrict__ src,
                                    __bf16* __restrict__ dst, int n) {
  int i = blockIdx.x * 256 + threadIdx.x;
  if (i < n) dst[i] = (__bf16)src[i];
}

// -------------------------------------------------------- embedding gather
__global__ void embed_kernel(const int* __restrict__ ids,
                             const float* __restrict__ table,
                             __bf16* __restrict__ emb16) {
  const int row = blockIdx.x;  // 0..ROWS-1
  const int id = ids[row];
  const float* src = table + (size_t)id * D_DIM;
  __bf16* dst = emb16 + (size_t)row * D_DIM;
  const int t = threadIdx.x;
  dst[t] = (__bf16)src[t];
  dst[t + 256] = (__bf16)src[t + 256];
}

// ------------------------------------------------- fused flash attention
// grid (S/64, H, B), block 128 (4 waves). Wave 0 DMAs each 32x512 K/V tile
// into double-buffered LDS (TENSORcnt pipelined); QK^T B-fragments and the
// V-transpose both read from LDS, so each tile crosses the fabric once.
__global__ __launch_bounds__(128, 1) void attn_kernel(
    const __bf16* __restrict__ emb, const unsigned char* __restrict__ pad,
    const float* __restrict__ self_factors,
    const float* __restrict__ dist_alphas, __bf16* __restrict__ ctx) {
  __shared__ __bf16 Kt[2][32][D_DIM];  // raw K/V tiles (2 x 32 KB)
  __shared__ __bf16 VT[D_DIM][32];     // transposed V tile (32 KB)
  __shared__ __bf16 Pst[4][16][32];    // per-wave P bounce (4 KB)

  const int lane = threadIdx.x & 31;
  const int wave = threadIdx.x >> 5;
  const int col = lane & 15;
  const int lh = lane >> 4;
  const int q0b = blockIdx.x * 64;
  const int q0 = q0b + wave * 16;
  const int h = blockIdx.y;
  const int b = blockIdx.z;
  const float alpha = dist_alphas[h];
  const float sfac = self_factors[h];
  const int win = (h == 1) ? 128 : (h == 2) ? 512 : S_LEN;
  const unsigned char* padrow = pad + b * S_LEN;
  const __bf16* E = emb + (size_t)b * S_LEN * D_DIM;
  const float inv_sqrt_d = 0.04419417382415922f;  // 1/sqrt(512)

  f32x8 acc[32];
#pragma unroll
  for (int i = 0; i < 32; ++i) acc[i] = f32x8_zero();
  float m[8], l[8];
#pragma unroll
  for (int i = 0; i < 8; ++i) { m[i] = -1e30f; l[i] = 0.0f; }

  int k_lo = q0b - win;
  if (k_lo < 0) k_lo = 0;
  k_lo &= ~31;
  const int T = (q0b + 64 - k_lo) / 32;

  if (wave == 0)
    tdm_load_2d(lds_off(&Kt[0][0][0]), E + (size_t)k_lo * D_DIM, D_DIM, 32,
                D_DIM);

  for (int t = 0; t < T; ++t) {
    const int kt = k_lo + t * 32;
    const int buf = t & 1;
    __syncthreads();  // iteration t-1 readers of Kt[buf^1]/VT are done
    if (wave == 0) {
      if (t + 1 < T) {
        tdm_load_2d(lds_off(&Kt[buf ^ 1][0][0]),
                    E + (size_t)(kt + 32) * D_DIM, D_DIM, 32, D_DIM);
        __builtin_amdgcn_s_wait_tensorcnt(1);  // tile t complete (in-order)
      } else {
        __builtin_amdgcn_s_wait_tensorcnt(0);
      }
    }
    __syncthreads();  // tile t visible to all waves

    // Stage transposed V tile from LDS: VT[d][kk] = Kt[buf][kk][d]
    for (int idx = threadIdx.x; idx < 16 * (D_DIM / 8); idx += 128) {
      const int kp = idx >> 6;        // key pair 0..15
      const int d8 = (idx & 63) * 8;  // d chunk
      bf16x8 v0 = *(const bf16x8*)&Kt[buf][2 * kp + 0][d8];
      bf16x8 v1 = *(const bf16x8*)&Kt[buf][2 * kp + 1][d8];
#pragma unroll
      for (int e = 0; e < 8; ++e) {
        bf16x2 pr;
        pr[0] = v0[e];
        pr[1] = v1[e];
        *(bf16x2*)&VT[d8 + e][2 * kp] = pr;
      }
    }
    __syncthreads();

    // scores: 16 q rows x 32 keys (Q frags hoisted to registers by compiler)
    f32x8 sfrag[2] = {f32x8_zero(), f32x8_zero()};
    for (int kd = 0; kd < D_DIM; kd += 32) {
      bf16x16 aq = load_a_frag(E + (size_t)q0 * D_DIM + kd, D_DIM, lane);
      bf16x16 bk0 = load_b_frag(&Kt[buf][0][kd], D_DIM, lane);
      bf16x16 bk1 = load_b_frag(&Kt[buf][16][kd], D_DIM, lane);
      sfrag[0] = wmma_bf16f32(aq, bk0, sfrag[0]);
      sfrag[1] = wmma_bf16f32(aq, bk1, sfrag[1]);
    }

    // decay / diag / masks (C-layout: row = i + 8*lh, col = kt + 16*f + col)
    float sc[2][8];
#pragma unroll
    for (int f = 0; f < 2; ++f)
#pragma unroll
      for (int i = 0; i < 8; ++i) {
        const int q = q0 + i + 8 * lh;
        const int k = kt + 16 * f + col;
        const int dist = q - k;
        const int adist = dist < 0 ? -dist : dist;
        float s = sfrag[f][i] * inv_sqrt_d;
        s *= 1.0f / (1.0f + alpha * (float)adist);
        if (dist == 0) s *= sfac;
        const bool kp = (padrow[k] != 0) && (dist != 0);
        const bool masked = (k > q) || (adist > win) || kp;
        sc[f][i] = masked ? -1e30f : s;
      }

    // online softmax per q row (row spread over 16 lanes of each half)
#pragma unroll
    for (int i = 0; i < 8; ++i) {
      float tm = fmaxf(sc[0][i], sc[1][i]);
#pragma unroll
      for (int off = 1; off < 16; off <<= 1)
        tm = fmaxf(tm, __shfl_xor(tm, off, 32));
      const float mn = fmaxf(m[i], tm);
      const float scale = __expf(m[i] - mn);
      float rs = 0.0f;
#pragma unroll
      for (int f = 0; f < 2; ++f) {
        float p = (sc[f][i] < -1e29f) ? 0.0f : __expf(sc[f][i] - mn);
        sc[f][i] = p;
        rs += p;
      }
#pragma unroll
      for (int off = 1; off < 16; off <<= 1) rs += __shfl_xor(rs, off, 32);
      l[i] = l[i] * scale + rs;
      m[i] = mn;
#pragma unroll
      for (int nf = 0; nf < 32; ++nf) acc[nf][i] *= scale;
    }

    // P (C-layout f32) -> LDS bf16 -> A-fragment layout
#pragma unroll
    for (int f = 0; f < 2; ++f)
#pragma unroll
      for (int i = 0; i < 8; ++i)
        Pst[wave][i + 8 * lh][16 * f + col] = (__bf16)sc[f][i];
    __syncthreads();
    bf16x16 pa = load_a_frag(&Pst[wave][0][0], 32, lane);

    // ctx += P @ V : B-fragments contiguous from transposed VT
#pragma unroll
    for (int nf = 0; nf < 32; ++nf) {
      bf16x16 bv = load_b_frag(&VT[nf * 16][0], 32, lane);
      acc[nf] = wmma_bf16f32(pa, bv, acc[nf]);
    }
  }

  // normalize and write straight into the concat layout [B,S,H*D]
#pragma unroll
  for (int i = 0; i < 8; ++i) {
    const float rl = 1.0f / l[i];
    const int q = q0 + i + 8 * lh;
    __bf16* dst = ctx + ((size_t)b * S_LEN + q) * HD + h * D_DIM;
#pragma unroll
    for (int nf = 0; nf < 32; ++nf)
      dst[nf * 16 + col] = (__bf16)(acc[nf][i] * rl);
  }
}

// --------------------------------------------------------- TDM-fed GEMM
// C[m][n] = sum_k A[m][k] * W[n][k] + bias[n]. Block 256 thr = 8 waves in a
// 4(M)x2(N) grid over a 128x128 tile. Wave 0 streams 128x64 A/B tiles into
// double-buffered LDS with tensor_load_to_lds; completion via TENSORcnt
// (in-order per wave: wait<=2 with next tile's two loads outstanding).
__global__ __launch_bounds__(256, 1) void gemm_bf16_kernel(
    const __bf16* __restrict__ A, const __bf16* __restrict__ W,
    const float* __restrict__ bias, float* __restrict__ Cf,
    __bf16* __restrict__ Cb, int M, int N, int K) {
  __shared__ __bf16 At[2][128][KT];  // 32 KB
  __shared__ __bf16 Bt[2][128][KT];  // 32 KB

  const int lane = threadIdx.x & 31;
  const int wave = threadIdx.x >> 5;
  const int col = lane & 15;
  const int lh = lane >> 4;
  const int mBlk = blockIdx.y * 128;
  const int nBlk = blockIdx.x * 128;
  const int m0 = (wave >> 1) * 32;  // within block tile
  const int n0 = (wave & 1) * 64;

  f32x8 acc[2][4];
#pragma unroll
  for (int i = 0; i < 2; ++i)
#pragma unroll
    for (int j = 0; j < 4; ++j) acc[i][j] = f32x8_zero();

  const int T = K / KT;
  if (wave == 0) {
    tdm_load_2d(lds_off(&At[0][0][0]), A + (size_t)mBlk * K, KT, 128, K);
    tdm_load_2d(lds_off(&Bt[0][0][0]), W + (size_t)nBlk * K, KT, 128, K);
  }

  for (int t = 0; t < T; ++t) {
    const int buf = t & 1;
    __syncthreads();  // readers of buf^1 (iteration t-1) are done
    if (wave == 0) {
      if (t + 1 < T) {
        tdm_load_2d(lds_off(&At[buf ^ 1][0][0]),
                    A + (size_t)mBlk * K + (size_t)(t + 1) * KT, KT, 128, K);
        tdm_load_2d(lds_off(&Bt[buf ^ 1][0][0]),
                    W + (size_t)nBlk * K + (size_t)(t + 1) * KT, KT, 128, K);
        __builtin_amdgcn_s_wait_tensorcnt(2);  // tile t complete (in-order)
      } else {
        __builtin_amdgcn_s_wait_tensorcnt(0);
      }
    }
    __syncthreads();  // tile t visible to all waves

#pragma unroll
    for (int ks = 0; ks < KT; ks += 32) {
      bf16x16 a0 = load_a_frag(&At[buf][m0 + 0][ks], KT, lane);
      bf16x16 a1 = load_a_frag(&At[buf][m0 + 16][ks], KT, lane);
      bf16x16 b0 = load_b_frag(&Bt[buf][n0 + 0][ks], KT, lane);
      bf16x16 b1 = load_b_frag(&Bt[buf][n0 + 16][ks], KT, lane);
      bf16x16 b2 = load_b_frag(&Bt[buf][n0 + 32][ks], KT, lane);
      bf16x16 b3 = load_b_frag(&Bt[buf][n0 + 48][ks], KT, lane);
      acc[0][0] = wmma_bf16f32(a0, b0, acc[0][0]);
      acc[0][1] = wmma_bf16f32(a0, b1, acc[0][1]);
      acc[0][2] = wmma_bf16f32(a0, b2, acc[0][2]);
      acc[0][3] = wmma_bf16f32(a0, b3, acc[0][3]);
      acc[1][0] = wmma_bf16f32(a1, b0, acc[1][0]);
      acc[1][1] = wmma_bf16f32(a1, b1, acc[1][1]);
      acc[1][2] = wmma_bf16f32(a1, b2, acc[1][2]);
      acc[1][3] = wmma_bf16f32(a1, b3, acc[1][3]);
    }
  }

#pragma unroll
  for (int j = 0; j < 4; ++j) {
    const int n = nBlk + n0 + 16 * j + col;
    const float bj = bias[n];
#pragma unroll
    for (int i = 0; i < 2; ++i)
#pragma unroll
      for (int e = 0; e < 8; ++e) {
        const float v = acc[i][j][e] + bj;
        const size_t idx = (size_t)(mBlk + m0 + 16 * i + 8 * lh + e) * N + n;
        if (Cf) Cf[idx] = v;
        else Cb[idx] = (__bf16)v;
      }
  }
}

extern "C" void kernel_launch(void* const* d_in, const int* in_sizes, int n_in,
                              void* d_out, int out_size, void* d_ws,
                              size_t ws_size, hipStream_t stream) {
  const int* ids = (const int*)d_in[0];
  const unsigned char* pad = (const unsigned char*)d_in[1];
  const float* table = (const float*)d_in[2];
  const float* sfac = (const float*)d_in[3];
  const float* dalpha = (const float*)d_in[4];
  const float* fw = (const float*)d_in[5];
  const float* fb = (const float*)d_in[6];
  const float* ow = (const float*)d_in[7];
  const float* ob = (const float*)d_in[8];
  float* out = (float*)d_out;

  char* w = (char*)d_ws;
  __bf16* emb16 = (__bf16*)w;   w += (size_t)ROWS * D_DIM * 2;   //  4 MB
  __bf16* ctx16 = (__bf16*)w;   w += (size_t)ROWS * HD * 2;      // 16 MB
  __bf16* fused16 = (__bf16*)w; w += (size_t)ROWS * D_DIM * 2;   //  4 MB
  __bf16* fw16 = (__bf16*)w;    w += (size_t)D_DIM * HD * 2;     //  2 MB
  __bf16* ow16 = (__bf16*)w;    w += (size_t)VOCAB * D_DIM * 2;  // 32 MB

  const int nfw = D_DIM * HD;     // 1,048,576
  const int now = VOCAB * D_DIM;  // 16,384,000
  cvt_f32_bf16_kernel<<<(nfw + 255) / 256, 256, 0, stream>>>(fw, fw16, nfw);
  cvt_f32_bf16_kernel<<<(now + 255) / 256, 256, 0, stream>>>(ow, ow16, now);
  embed_kernel<<<ROWS, 256, 0, stream>>>(ids, table, emb16);
  attn_kernel<<<dim3(S_LEN / 64, NHEAD, BATCH), 128, 0, stream>>>(
      emb16, pad, sfac, dalpha, ctx16);
  // fused = concat @ fusion_w^T + fb : M=4096, N=512, K=2048
  gemm_bf16_kernel<<<dim3(D_DIM / 128, ROWS / 128), 256, 0, stream>>>(
      ctx16, fw16, fb, nullptr, fused16, ROWS, D_DIM, HD);
  // logits = fused @ out_w^T + ob : M=4096, N=32000, K=512
  gemm_bf16_kernel<<<dim3(VOCAB / 128, ROWS / 128), 256, 0, stream>>>(
      fused16, ow16, ob, out, nullptr, ROWS, VOCAB, D_DIM);
}
